// ReduLayer_N_X_M_10840497455778
// MI455X (gfx1250) — compile-verified, matrix operand order checked by reading, not run on hardware
//
#include <hip/hip_runtime.h>

// ---------------------------------------------------------------------------
// Problem constants (match reference)
// ---------------------------------------------------------------------------
constexpr int B  = 32;
constexpr int K  = 128;
constexpr int N  = 128;   // n_dim == num_classes
constexpr int M  = 256;   // samples per matrix
constexpr int NN = N * N;         // 16384
constexpr int NM = N * M;         // 32768
constexpr int BNM = B * N * M;    // 1048576
constexpr float EPS2      = 0.01f;   // EPSILON^2
constexpr float LAMBDA_SM = 0.1f;
constexpr float LR        = 0.01f;
constexpr float BN_EPS    = 1e-5f;

// ---------------------------------------------------------------------------
// WMMA / TDM vector types (wave32, CDNA5)
// ---------------------------------------------------------------------------
typedef __attribute__((ext_vector_type(16))) __bf16 v16bf;
typedef __attribute__((ext_vector_type(8)))  __bf16 v8bf;
typedef __attribute__((ext_vector_type(8)))  float  v8f;
typedef __attribute__((ext_vector_type(4)))  unsigned int u32x4;
typedef __attribute__((ext_vector_type(8)))  int          i32x8;
typedef __attribute__((ext_vector_type(4)))  int          i32x4;

// Load a 16-element bf16 fragment half-pair: 8 contiguous bf16 at p, 8 at p+16.
// Matches the 16-bit A/B operand layout (lane = row/col, VGPR quads hold
// 8 consecutive K at offsets khalf*8 and 16+khalf*8).
__device__ __forceinline__ v16bf load_frag(const __bf16* p) {
    v8bf lo = *reinterpret_cast<const v8bf*>(p);
    v8bf hi = *reinterpret_cast<const v8bf*>(p + 16);
    return __builtin_shufflevector(lo, hi, 0,1,2,3,4,5,6,7,8,9,10,11,12,13,14,15);
}

// ---------------------------------------------------------------------------
// Tensor Data Mover: DMA a rows x cols bf16 tile (row stride = strideElems)
// from global memory into LDS. Issued by ONE wave; EXEC is ignored by TDM.
// D# layout per CDNA5 ISA ch.8: group0 = {flags, lds_addr, global_addr, type},
// group1 = {data_size, tensor dims/strides, tile dims}; groups 2/3 zero (2-D).
// This toolchain exposes the 6-arg builtin (extra int32x8 group + cpol).
// ---------------------------------------------------------------------------
__device__ __forceinline__ void tdm_load_tile_bf16(const void* gsrc, void* lds_dst,
                                                   unsigned rows, unsigned cols,
                                                   unsigned strideElems) {
#if __has_builtin(__builtin_amdgcn_tensor_load_to_lds)
    unsigned long long ga = (unsigned long long)(uintptr_t)gsrc;
    unsigned lds_off = (unsigned)(uintptr_t)lds_dst;   // LDS aperture: low 32 bits = offset
    u32x4 g0;
    g0.x = 1u;                                              // count=1 valid user descriptor
    g0.y = lds_off;                                         // lds_addr (bytes)
    g0.z = (unsigned)(ga & 0xFFFFFFFFu);                    // global_addr[31:0]
    g0.w = (unsigned)((ga >> 32) & 0x1FFFFFFu) | (2u << 30);// global_addr[56:32] | type=2
    i32x8 g1;
    g1[0] = (int)(1u << 16);                                // data_size=1 (2 bytes/elem)
    g1[1] = (int)((cols & 0xFFFFu) << 16);                  // tensor_dim0[15:0]
    g1[2] = (int)((cols >> 16) | ((rows & 0xFFFFu) << 16)); // tensor_dim0 hi | tensor_dim1 lo
    g1[3] = (int)((rows >> 16) | ((cols & 0xFFFFu) << 16)); // tensor_dim1 hi | tile_dim0
    g1[4] = (int)(rows & 0xFFFFu);                          // tile_dim1 | tile_dim2=0
    g1[5] = (int)strideElems;                               // tensor_dim0_stride[31:0]
    g1[6] = 0;                                              // stride0 hi | stride1 lo
    g1[7] = 0;                                              // stride1 hi
    i32x4 z4 = {0, 0, 0, 0};
    i32x8 z8 = {0, 0, 0, 0, 0, 0, 0, 0};
    __builtin_amdgcn_tensor_load_to_lds(g0, g1, z4, z4, z8, 0);
    __builtin_amdgcn_s_wait_tensorcnt((short)0);
#else
    (void)gsrc; (void)lds_dst; (void)rows; (void)cols; (void)strideElems;
#endif
}

// Stage a rows x cols bf16 tile into LDS: TDM path (wave 0) or cooperative copy.
__device__ __forceinline__ void stage_tile_bf16(const __bf16* gsrc, __bf16* lds_dst,
                                                unsigned rows, unsigned cols,
                                                unsigned strideElems, int nthreads) {
#if __has_builtin(__builtin_amdgcn_tensor_load_to_lds)
    if (threadIdx.x < 32u)
        tdm_load_tile_bf16(gsrc, lds_dst, rows, cols, strideElems);
#else
    const int nelem = (int)(rows * cols) / 8;
    for (int i = threadIdx.x; i < nelem; i += nthreads)
        reinterpret_cast<v8bf*>(lds_dst)[i] =
            reinterpret_cast<const v8bf*>(gsrc)[i];   // valid when cols == strideElems
#endif
    __syncthreads();
}

// ---------------------------------------------------------------------------
// 1) BatchNorm statistics: per-channel mean / rstd over (B, M)
// ---------------------------------------------------------------------------
__global__ __launch_bounds__(256)
void bn_stats_kernel(const float* __restrict__ x,
                     float* __restrict__ meanbuf, float* __restrict__ rstdbuf) {
    const int n = blockIdx.x;
    const int tid = threadIdx.x;
    float s = 0.f, sq = 0.f;
    for (int i = tid; i < B * M; i += 256) {
        int b = i / M, m = i - b * M;
        float v = x[(size_t)b * NM + (size_t)n * M + m];
        s += v; sq += v * v;
    }
    __shared__ float r1[256], r2[256];
    r1[tid] = s; r2[tid] = sq; __syncthreads();
    for (int o = 128; o > 0; o >>= 1) {
        if (tid < o) { r1[tid] += r1[tid + o]; r2[tid] += r2[tid + o]; }
        __syncthreads();
    }
    if (tid == 0) {
        float mean = r1[0] * (1.f / (B * M));
        float var  = r2[0] * (1.f / (B * M)) - mean * mean;  // biased
        meanbuf[n] = mean;
        rstdbuf[n] = rsqrtf(var + BN_EPS);
    }
}

// ---------------------------------------------------------------------------
// 2) Apply BN: write x̂ (f32, (B,N,M)) and x̂ᵀ (bf16, (B,M,N)) for WMMA B-op
// ---------------------------------------------------------------------------
__global__ __launch_bounds__(256)
void bn_apply_kernel(const float* __restrict__ x,
                     const float* __restrict__ meanbuf, const float* __restrict__ rstdbuf,
                     float* __restrict__ xhat, __bf16* __restrict__ xhatT) {
    size_t idx = (size_t)blockIdx.x * 256 + threadIdx.x;   // < BNM
    int m = (int)(idx % M);
    size_t t = idx / M;
    int n = (int)(t % N);
    int b = (int)(t / N);
    float v = (x[idx] - meanbuf[n]) * rstdbuf[n];
    xhat[idx] = v;
    xhatT[(size_t)b * NM + (size_t)m * N + n] = (__bf16)v;
}

// ---------------------------------------------------------------------------
// 3) total = Σ EC_count, e_alpha, has_counts flag
// ---------------------------------------------------------------------------
__global__ __launch_bounds__(128)
void count_total_kernel(const float* __restrict__ EC_count, float* __restrict__ scalars) {
    int tid = threadIdx.x;
    __shared__ float r[128];
    r[tid] = EC_count[tid]; __syncthreads();
    for (int o = 64; o > 0; o >>= 1) { if (tid < o) r[tid] += r[tid + o]; __syncthreads(); }
    if (tid == 0) {
        float total = r[0];
        bool has = total > 1e-6f;
        scalars[0] = total;
        scalars[1] = has ? (float)N / (EPS2 * total) : 0.f;   // e_alpha
        scalars[2] = has ? 1.f : 0.f;
    }
}

// ---------------------------------------------------------------------------
// 4) Psum = Σ_k EC_proto[k]
// ---------------------------------------------------------------------------
__global__ __launch_bounds__(256)
void proto_sum_kernel(const float* __restrict__ EC_proto, float* __restrict__ Psum) {
    int e = blockIdx.x * 256 + threadIdx.x;   // < NN
    float s = 0.f;
    for (int k = 0; k < K; ++k) s += EC_proto[(size_t)k * NN + e];
    Psum[e] = s;
}

// ---------------------------------------------------------------------------
// 5) Invert 129 SPD matrices (αP + I) via in-place Gauss-Jordan in LDS
//    (SPD ⇒ no pivoting required). Output α·inv(...) as f32 C_k + bf16 C_k, or E.
//    LDS row stride 129 → bank-conflict-free column sweeps.
// ---------------------------------------------------------------------------
__global__ __launch_bounds__(128)
void invert_kernel(const float* __restrict__ EC_proto, const float* __restrict__ EC_count,
                   const float* __restrict__ Psum, const float* __restrict__ scalars,
                   float* __restrict__ Cf, __bf16* __restrict__ Cbf, float* __restrict__ Ef) {
    const int g = blockIdx.x;      // 0..K-1 => class matrices, K => global E
    const int tid = threadIdx.x;   // row ownership
    float alpha;
    const float* src;
    if (g < K) {
        float cnt = EC_count[g];
        float bd  = (cnt != 0.f) ? 1.f : 0.f;
        float fix = cnt + (1.f - bd);
        alpha = ((float)N / (fix * EPS2)) * bd;
        src = EC_proto + (size_t)g * NN;
    } else {
        alpha = scalars[1];
        src = Psum;
    }

    __shared__ float S[128 * 129];
    for (int c = 0; c < 128; ++c)
        S[tid * 129 + c] = alpha * src[tid * 128 + c] + ((c == tid) ? 1.f : 0.f);

    for (int j = 0; j < 128; ++j) {
        __syncthreads();
        if (tid == j) {
            float pinv = 1.f / S[j * 129 + j];
            S[j * 129 + j] = 1.f;
            for (int c = 0; c < 128; ++c) S[j * 129 + c] *= pinv;
        }
        __syncthreads();
        if (tid != j) {
            float f = S[tid * 129 + j];
            S[tid * 129 + j] = 0.f;
            for (int c = 0; c < 128; ++c) S[tid * 129 + c] -= f * S[j * 129 + c];
        }
    }
    __syncthreads();

    if (g < K) {
        for (int c = 0; c < 128; ++c) {
            float v = alpha * S[tid * 129 + c];
            Cf [(size_t)g * NN + tid * 128 + c] = v;
            Cbf[(size_t)g * NN + tid * 128 + c] = (__bf16)v;
        }
    } else {
        for (int c = 0; c < 128; ++c) Ef[tid * 128 + c] = alpha * S[tid * 129 + c];
    }
}

// ---------------------------------------------------------------------------
// 6) Heavy pass: ||C_k @ X_b||_F^2 via bf16 WMMA.
//    C_k (32 KB) staged once per workgroup into LDS via the Tensor Data Mover;
//    A-fragments then come from ds_load_b128 instead of 8x-redundant VMEM.
//    B strip (2x4 fragments, 64 VGPRs) held resident across the ti loop.
// ---------------------------------------------------------------------------
__global__ __launch_bounds__(256)
void norm_gemm_kernel(const __bf16* __restrict__ Cbf, const __bf16* __restrict__ xhatT,
                      float* __restrict__ normbuf) {
    const int b = blockIdx.x >> 7;
    const int k = blockIdx.x & 127;
    const int wave = threadIdx.x >> 5;
    const int lane = threadIdx.x & 31;
    const int lm = lane & 15, kh = lane >> 4;

    __shared__ __align__(16) __bf16 ldsC[NN];              // 32 KB staged C_k
    const __bf16* Ak = Cbf   + (size_t)k * NN;
    const __bf16* Bb = xhatT + (size_t)b * NM;

    stage_tile_bf16(Ak, ldsC, N, N, N, 256);               // TDM + s_wait_tensorcnt

    // Preload the wave's B fragments (held in VGPRs across all row tiles).
    v16bf bfr[2][4];
#pragma unroll
    for (int tj2 = 0; tj2 < 2; ++tj2) {
        const int tj = wave * 2 + tj2;
        const __bf16* brow = Bb + (size_t)(tj * 16 + lm) * N + kh * 8;
#pragma unroll
        for (int kk = 0; kk < 4; ++kk) bfr[tj2][kk] = load_frag(brow + kk * 32);
    }

    float sumsq = 0.f;
    for (int ti = 0; ti < 8; ++ti) {                       // 8 row tiles of 16 (N)
        const __bf16* arow = ldsC + (ti * 16 + lm) * N + kh * 8;
        v16bf a[4];
#pragma unroll
        for (int kk = 0; kk < 4; ++kk) a[kk] = load_frag(arow + kk * 32);
#pragma unroll
        for (int tj2 = 0; tj2 < 2; ++tj2) {
            v8f acc = {};
#pragma unroll
            for (int kk = 0; kk < 4; ++kk)                 // K = 128 = 4 x 32
                acc = __builtin_amdgcn_wmma_f32_16x16x32_bf16(
                        false, a[kk], false, bfr[tj2][kk], (short)0, acc, false, false);
#pragma unroll
            for (int i = 0; i < 8; ++i) sumsq += acc[i] * acc[i];
        }
    }
    for (int off = 16; off > 0; off >>= 1) sumsq += __shfl_down(sumsq, off, 32);
    if (lane == 0) atomicAdd(&normbuf[b * K + k], sumsq);
}

// ---------------------------------------------------------------------------
// 7) Norms -> rescale(Σ=10) -> softmax(-λ·) -> *bool_depth -> w = γ·π
// ---------------------------------------------------------------------------
__global__ __launch_bounds__(128)
void pi_weights_kernel(const float* __restrict__ normbuf, const float* __restrict__ EC_count,
                       const float* __restrict__ scalars, float* __restrict__ wgt) {
    const int b = blockIdx.x, k = threadIdx.x;
    __shared__ float red[128];
    float nb = sqrtf(fmaxf(normbuf[b * K + k], 0.f));
    red[k] = nb; __syncthreads();
    for (int o = 64; o > 0; o >>= 1) { if (k < o) red[k] += red[k + o]; __syncthreads(); }
    float s = red[0]; __syncthreads();
    float nr = 10.f * nb / s;
    float p = expf(-LAMBDA_SM * nr);
    red[k] = p; __syncthreads();
    for (int o = 64; o > 0; o >>= 1) { if (k < o) red[k] += red[k + o]; __syncthreads(); }
    float sp = red[0];
    float cnt = EC_count[k];
    float bd  = (cnt != 0.f) ? 1.f : 0.f;
    float pi  = (p / sp) * bd;
    float gama = (scalars[2] != 0.f) ? cnt / scalars[0] : 0.f;
    wgt[b * K + k] = gama * pi;
}

// ---------------------------------------------------------------------------
// 8) W_b = E - Σ_k w_bk C_k   (linearity: avoids recomputing 34 GF pass)
// ---------------------------------------------------------------------------
__global__ __launch_bounds__(256)
void build_w_kernel(const float* __restrict__ Ef, const float* __restrict__ Cf,
                    const float* __restrict__ wgt, __bf16* __restrict__ Wbf) {
    size_t flat = (size_t)blockIdx.x * 256 + threadIdx.x;   // < B*NN
    int e = (int)(flat & (NN - 1));
    int b = (int)(flat >> 14);
    float acc = Ef[e];
    const float* wrow = wgt + b * K;
    for (int k = 0; k < K; ++k) {
        if ((k & 15) == 0 && k + 16 < K)
            __builtin_prefetch(&Cf[(size_t)(k + 16) * NN + e], 0, 0);
        acc -= wrow[k] * Cf[(size_t)k * NN + e];
    }
    Wbf[flat] = (__bf16)acc;
}

// ---------------------------------------------------------------------------
// 9) out[b] = x̂[b] + LR · (W_b @ X̂_b)   via bf16 WMMA, W_b staged via TDM
// ---------------------------------------------------------------------------
__global__ __launch_bounds__(256)
void final_gemm_kernel(const __bf16* __restrict__ Wbf, const __bf16* __restrict__ xhatT,
                       const float* __restrict__ xhat, float* __restrict__ out) {
    const int b = blockIdx.x;
    const int wave = threadIdx.x >> 5;
    const int lane = threadIdx.x & 31;
    const int lm = lane & 15, kh = lane >> 4;

    __shared__ __align__(16) __bf16 ldsW[NN];              // 32 KB staged W_b
    const __bf16* Ab = Wbf   + (size_t)b * NN;
    const __bf16* Bb = xhatT + (size_t)b * NM;

    stage_tile_bf16(Ab, ldsW, N, N, N, 256);

    v16bf bfr[2][4];
#pragma unroll
    for (int tj2 = 0; tj2 < 2; ++tj2) {
        const int tj = wave * 2 + tj2;
        const __bf16* brow = Bb + (size_t)(tj * 16 + lm) * N + kh * 8;
#pragma unroll
        for (int kk = 0; kk < 4; ++kk) bfr[tj2][kk] = load_frag(brow + kk * 32);
    }

    for (int ti = 0; ti < 8; ++ti) {
        const __bf16* arow = ldsW + (ti * 16 + lm) * N + kh * 8;
        v16bf a[4];
#pragma unroll
        for (int kk = 0; kk < 4; ++kk) a[kk] = load_frag(arow + kk * 32);
#pragma unroll
        for (int tj2 = 0; tj2 < 2; ++tj2) {
            const int tj = wave * 2 + tj2;
            v8f acc = {};
#pragma unroll
            for (int kk = 0; kk < 4; ++kk)
                acc = __builtin_amdgcn_wmma_f32_16x16x32_bf16(
                        false, a[kk], false, bfr[tj2][kk], (short)0, acc, false, false);
            // C/D layout: reg r, lanes 0-15 -> row r, lanes 16-31 -> row 8+r
            const int row0 = ti * 16 + kh * 8;
            const int col  = tj * 16 + lm;
#pragma unroll
            for (int r = 0; r < 8; ++r) {
                size_t o = (size_t)b * NM + (size_t)(row0 + r) * M + col;
                out[o] = xhat[o] + LR * acc[r];
            }
        }
    }
}

// ---------------------------------------------------------------------------
// Host-side launch
// ---------------------------------------------------------------------------
extern "C" void kernel_launch(void* const* d_in, const int* in_sizes, int n_in,
                              void* d_out, int out_size, void* d_ws, size_t ws_size,
                              hipStream_t stream) {
    (void)in_sizes; (void)n_in; (void)out_size; (void)ws_size;

    const float* x        = (const float*)d_in[0];   // (B,N,M)
    const float* EC_proto = (const float*)d_in[1];   // (K,N,N)
    const float* EC_count = (const float*)d_in[2];   // (K,)
    float* out = (float*)d_out;

    char* ws = (char*)d_ws;
    size_t off = 0;
    auto alloc = [&](size_t bytes) -> void* {
        void* p = ws + off;
        off += (bytes + 255) & ~(size_t)255;
        return p;
    };
    float*  xhat    = (float*)alloc(sizeof(float) * BNM);            // 4 MB
    __bf16* xhatT   = (__bf16*)alloc(2ull * BNM);                    // 2 MB
    float*  Cf      = (float*)alloc(sizeof(float) * (size_t)K * NN); // 8 MB
    __bf16* Cbf     = (__bf16*)alloc(2ull * (size_t)K * NN);         // 4 MB
    float*  Ef      = (float*)alloc(sizeof(float) * NN);
    float*  Psum    = (float*)alloc(sizeof(float) * NN);
    __bf16* Wbf     = (__bf16*)alloc(2ull * (size_t)B * NN);         // 1 MB
    float*  normbuf = (float*)alloc(sizeof(float) * B * K);
    float*  wgt     = (float*)alloc(sizeof(float) * B * K);
    float*  meanbuf = (float*)alloc(sizeof(float) * N);
    float*  rstdbuf = (float*)alloc(sizeof(float) * N);
    float*  scalars = (float*)alloc(sizeof(float) * 8);

    bn_stats_kernel  <<<N, 256, 0, stream>>>(x, meanbuf, rstdbuf);
    bn_apply_kernel  <<<BNM / 256, 256, 0, stream>>>(x, meanbuf, rstdbuf, xhat, xhatT);
    count_total_kernel<<<1, 128, 0, stream>>>(EC_count, scalars);
    proto_sum_kernel <<<NN / 256, 256, 0, stream>>>(EC_proto, Psum);
    invert_kernel    <<<K + 1, 128, 0, stream>>>(EC_proto, EC_count, Psum, scalars, Cf, Cbf, Ef);
    (void)hipMemsetAsync(normbuf, 0, sizeof(float) * B * K, stream);
    norm_gemm_kernel <<<B * K, 256, 0, stream>>>(Cbf, xhatT, normbuf);
    pi_weights_kernel<<<B, 128, 0, stream>>>(normbuf, EC_count, scalars, wgt);
    build_w_kernel   <<<(B * NN) / 256, 256, 0, stream>>>(Ef, Cf, wgt, Wbf);
    final_gemm_kernel<<<B, 256, 0, stream>>>(Wbf, xhatT, xhat, out);
}
